// AnyBCQLinear_87814901334608
// MI455X (gfx1250) — matrix operand order repaired and checked by reading, not compiled, not run on hardware
//
#include <hip/hip_runtime.h>

// ---------------------------------------------------------------------------
// BCQ linear: y[M,N] = x[M,K] @ W[K,N],  M=8192, K=4096, N=4096
//  k0: x f32 -> bf16 (once, if ws fits)       [compute-bound problem: 275 GFLOP
//  k1: dequant W -> bf16 tiled ws              vs ~300MB traffic => feed the
//  k2: bf16 WMMA GEMM, f32 accum, double-      bf16 matrix pipes, stage via
//      buffered LDS + async global->LDS        async-to-LDS, hide everything]
// ---------------------------------------------------------------------------

#define IN_F   4096
#define OUT_F  4096
#define M_ROWS 8192
#define KBLKS  128         // IN_F / 32

typedef __attribute__((ext_vector_type(16))) __bf16 v16bf;
typedef __attribute__((ext_vector_type(8)))  __bf16 v8bf;
typedef __attribute__((ext_vector_type(8)))  float  v8f;
typedef int v4i_vs __attribute__((vector_size(16)));   // matches builtin param

// ---- optional CDNA5 async global->LDS path --------------------------------
#if defined(__has_builtin)
#if __has_builtin(__builtin_amdgcn_global_load_async_to_lds_b128)
#define HAS_ASYNC_LDS 1
#endif
#if __has_builtin(__builtin_amdgcn_s_wait_asynccnt)
#define HAS_WAIT_ASYNC 1
#endif
#endif

// generic->AS casts via integer (global: flat address == AS1 address bitwise;
// LDS: flat LDS addresses carry the 32-bit LDS offset in addr[31:0], ISA 10.2)
#define GLB_V4(p) ((__attribute__((address_space(1))) v4i_vs*)(uintptr_t)(p))
#define LDS_V4(p) ((__attribute__((address_space(3))) v4i_vs*)(unsigned)(uintptr_t)(p))

__device__ __forceinline__ void wait_async0() {
#if defined(HAS_ASYNC_LDS)
#if defined(HAS_WAIT_ASYNC)
    __builtin_amdgcn_s_wait_asynccnt(0);
#else
    asm volatile("s_wait_asynccnt 0x0" ::: "memory");
#endif
#endif
}

__device__ __forceinline__ unsigned short f2bf(float f) {
    unsigned u = __float_as_uint(f);
    u += 0x7fffu + ((u >> 16) & 1u);   // round-to-nearest-even
    return (unsigned short)(u >> 16);
}
__device__ __forceinline__ unsigned pack2bf(float lo, float hi) {
    return (unsigned)f2bf(lo) | ((unsigned)f2bf(hi) << 16);
}

// ---------------------------------------------------------------------------
// Kernel 0: x f32 -> bf16, row-major copy (one time)
// ---------------------------------------------------------------------------
__global__ __launch_bounds__(256)
void x_cvt_kernel(const float* __restrict__ X, unsigned* __restrict__ X16) {
    const size_t base = ((size_t)blockIdx.x * 256 + threadIdx.x) * 8;
    float4 a = *(const float4*)(X + base);
    float4 b = *(const float4*)(X + base + 4);
    uint4 o;
    o.x = pack2bf(a.x, a.y);
    o.y = pack2bf(a.z, a.w);
    o.z = pack2bf(b.x, b.y);
    o.w = pack2bf(b.z, b.w);
    *(uint4*)(X16 + base / 2) = o;
}

// ---------------------------------------------------------------------------
// Kernel 1: dequantize W -> bf16, tiled ws[kblk][o][k%32]
// => every (32k x 128n) B-tile is one contiguous 8KB block.
// ---------------------------------------------------------------------------
__global__ __launch_bounds__(256)
void bcq_dequant_kernel(const int* __restrict__ qw,
                        const float* __restrict__ alpha,
                        const float* __restrict__ beta,
                        unsigned* __restrict__ wsB)
{
    const int o   = blockIdx.y * 256 + threadIdx.x;   // 0..4095
    const int blk = blockIdx.x;                       // 0..127 (k/32)
    const int g   = blk >> 2;                         // 0..31  (k/128)

    const int   q0 = qw[(blk * 4 + 0) * OUT_F + o];
    const int   q1 = qw[(blk * 4 + 1) * OUT_F + o];
    const int   q2 = qw[(blk * 4 + 2) * OUT_F + o];
    const int   q3 = qw[(blk * 4 + 3) * OUT_F + o];
    const float a0 = alpha[(g * 4 + 0) * OUT_F + o];
    const float a1 = alpha[(g * 4 + 1) * OUT_F + o];
    const float a2 = alpha[(g * 4 + 2) * OUT_F + o];
    const float a3 = alpha[(g * 4 + 3) * OUT_F + o];
    const float bt = beta[g * OUT_F + o];

    unsigned packed[16];
#pragma unroll
    for (int sp = 0; sp < 16; ++sp) {
        const int s0 = 2 * sp, s1 = 2 * sp + 1;
        float w0 = bt, w1 = bt;
        w0 += ((q0 >> s0) & 1) ? a0 : -a0;
        w0 += ((q1 >> s0) & 1) ? a1 : -a1;
        w0 += ((q2 >> s0) & 1) ? a2 : -a2;
        w0 += ((q3 >> s0) & 1) ? a3 : -a3;
        w1 += ((q0 >> s1) & 1) ? a0 : -a0;
        w1 += ((q1 >> s1) & 1) ? a1 : -a1;
        w1 += ((q2 >> s1) & 1) ? a2 : -a2;
        w1 += ((q3 >> s1) & 1) ? a3 : -a3;
        packed[sp] = pack2bf(w0, w1);
    }

    uint4* dst = (uint4*)(wsB + ((size_t)blk * OUT_F + o) * 16);
#pragma unroll
    for (int j = 0; j < 4; ++j) {
        uint4 v;
        v.x = packed[4 * j + 0];
        v.y = packed[4 * j + 1];
        v.z = packed[4 * j + 2];
        v.w = packed[4 * j + 3];
        dst[j] = v;
    }
}

// ---------------------------------------------------------------------------
// Kernel 2: GEMM via v_wmma_f32_16x16x32_bf16
// Block tile 128x128, K-step 64 (16 WMMA between barriers), 8 waves, wave
// tile 32x64 (2x4 accum). Double-buffered LDS; tile k+1 prefetched (async
// global->LDS when available) while tile k is computed.
// ---------------------------------------------------------------------------
#define LD   72            // padded LDS row stride (bf16): 144B = 36 banks,
                           // frag b128 starts land on banks 0,4,...,60: clean
#define ATILE (128 * LD)   // elems per buffer

template <int ABF16>
__global__ __launch_bounds__(256)
void bcq_gemm_kernel(const float*  __restrict__ Xf,    // used if !ABF16
                     const __bf16* __restrict__ X16,   // used if ABF16
                     const __bf16* __restrict__ Wt,    // ws[kblk][o][k%32]
                     float* __restrict__ Y)
{
    __shared__ __bf16 As[2 * ATILE];
    __shared__ __bf16 Bs[2 * ATILE];

    const int tid  = threadIdx.x;
    const int lane = tid & 31;
    const int wave = tid >> 5;
    const int wm   = wave & 3;        // row offset wm*32
    const int wn   = wave >> 2;       // col offset wn*64
    const int l15  = lane & 15;
    const int lhi  = lane >> 4;       // which K-half this lane holds

    const int gm0 = blockIdx.x * 128;
    const int gn0 = blockIdx.y * 128;

    v8f acc[2][4];
#pragma unroll
    for (int m = 0; m < 2; ++m)
#pragma unroll
        for (int n = 0; n < 4; ++n)
            acc[m][n] = (v8f)0.0f;

    // ---- staging: A tile 128x64 bf16 ---------------------------------------
    auto stageA = [&](int kb, __bf16* dst) {
        if (ABF16) {
#if defined(HAS_ASYNC_LDS)
#pragma unroll
            for (int it = 0; it < 4; ++it) {
                const int u = it * 256 + tid;        // 128 rows x 8 segs(16B)
                const int row = u >> 3, seg = (u & 7) * 8;
                __builtin_amdgcn_global_load_async_to_lds_b128(
                    GLB_V4(X16 + (size_t)(gm0 + row) * IN_F + kb * 64 + seg),
                    LDS_V4(dst + row * LD + seg), 0, 0);
            }
#else
#pragma unroll
            for (int it = 0; it < 4; ++it) {
                const int u = it * 256 + tid;
                const int row = u >> 3, seg = (u & 7) * 8;
                uint4 v = *(const uint4*)(X16 + (size_t)(gm0 + row) * IN_F + kb * 64 + seg);
                *(uint4*)&dst[row * LD + seg] = v;
            }
#endif
        } else {
#pragma unroll
            for (int it = 0; it < 8; ++it) {
                const int u = it * 256 + tid;        // 128 rows x 16 segs(4 f32)
                const int row = u >> 4, col = (u & 15) * 4;
                float4 v = *(const float4*)(Xf + (size_t)(gm0 + row) * IN_F + kb * 64 + col);
                uint2 p;
                p.x = pack2bf(v.x, v.y);
                p.y = pack2bf(v.z, v.w);
                *(uint2*)&dst[row * LD + col] = p;
            }
        }
    };

    // ---- staging: B tile 128n x 64k bf16 (two contiguous 8KB blocks) -------
    auto stageB = [&](int kb, __bf16* dst) {
#pragma unroll
        for (int c = 0; c < 2; ++c) {
            const __bf16* src = Wt + (size_t)(kb * 2 + c) * (OUT_F * 32) + (size_t)gn0 * 32;
#pragma unroll
            for (int it = 0; it < 2; ++it) {
                const int u = it * 256 + tid;        // 512 x 16B
                const int n = u >> 2, q = (u & 3) * 8;
#if defined(HAS_ASYNC_LDS)
                __builtin_amdgcn_global_load_async_to_lds_b128(
                    GLB_V4(src + (size_t)u * 8),
                    LDS_V4(dst + n * LD + c * 32 + q), 0, 0);
#else
                uint4 v = *(const uint4*)(src + (size_t)u * 8);
                *(uint4*)&dst[n * LD + c * 32 + q] = v;
#endif
            }
        }
    };

    // ---- compute one 64-wide K step from LDS buffers ------------------------
    auto compute = [&](const __bf16* A, const __bf16* B) {
#pragma unroll
        for (int c = 0; c < 2; ++c) {
            v16bf af[2], bfr[4];
            const int kA = c * 32 + lhi * 16;  // A: lanes0-15 K0..15, 16-31 K16..31
#pragma unroll
            for (int m = 0; m < 2; ++m) {
                const __bf16* p = &A[(wm * 32 + m * 16 + l15) * LD + kA];
                union { v16bf v; v8bf h[2]; } u;
                u.h[0] = *(const v8bf*)(p);
                u.h[1] = *(const v8bf*)(p + 8);
                af[m] = u.v;
            }
            const int kB = c * 32 + lhi * 16;  // B: lanes0-15 K0..15, 16-31 K16..31
#pragma unroll
            for (int n = 0; n < 4; ++n) {
                const __bf16* p = &B[(wn * 64 + n * 16 + l15) * LD + kB];
                union { v16bf v; v8bf h[2]; } u;
                u.h[0] = *(const v8bf*)(p);
                u.h[1] = *(const v8bf*)(p + 8);
                bfr[n] = u.v;
            }
#pragma unroll
            for (int m = 0; m < 2; ++m)
#pragma unroll
                for (int n = 0; n < 4; ++n)
                    acc[m][n] = __builtin_amdgcn_wmma_f32_16x16x32_bf16(
                        false, af[m], false, bfr[n], (short)0, acc[m][n],
                        false, false);
        }
    };

    // ---- software pipeline --------------------------------------------------
    const int NK = IN_F / 64;          // 64 iterations
    stageA(0, As);
    stageB(0, Bs);
    wait_async0();
    __syncthreads();

    for (int kb = 0; kb < NK; ++kb) {
        const int cur = kb & 1;
        if (kb + 1 < NK) {             // prefetch next tile into other buffer
            stageA(kb + 1, As + (cur ^ 1) * ATILE);
            stageB(kb + 1, Bs + (cur ^ 1) * ATILE);
        }
        compute(As + cur * ATILE, Bs + cur * ATILE);
        wait_async0();
        __syncthreads();
    }

    // ---- epilogue: C/D layout: VGPR r -> (M = r + lhi*8, N = l15) ----------
#pragma unroll
    for (int m = 0; m < 2; ++m) {
#pragma unroll
        for (int n = 0; n < 4; ++n) {
            const int row0 = gm0 + wm * 32 + m * 16 + lhi * 8;
            const int col  = gn0 + wn * 64 + n * 16 + l15;
#pragma unroll
            for (int r = 0; r < 8; ++r)
                Y[(size_t)(row0 + r) * OUT_F + col] = acc[m][n][r];
        }
    }
}

// ---------------------------------------------------------------------------
extern "C" void kernel_launch(void* const* d_in, const int* in_sizes, int n_in,
                              void* d_out, int out_size, void* d_ws, size_t ws_size,
                              hipStream_t stream) {
    const float* x     = (const float*)d_in[0];
    const int*   qw    = (const int*)d_in[1];
    const float* alpha = (const float*)d_in[2];
    const float* beta  = (const float*)d_in[3];
    float*       y     = (float*)d_out;

    const size_t X16_BYTES = (size_t)M_ROWS * IN_F * 2;   // 64 MiB
    const size_t WB_BYTES  = (size_t)IN_F * OUT_F * 2;    // 32 MiB
    const bool   big       = ws_size >= X16_BYTES + WB_BYTES;

    unsigned* x16 = (unsigned*)d_ws;
    unsigned* wb  = big ? (unsigned*)((char*)d_ws + X16_BYTES) : (unsigned*)d_ws;

    bcq_dequant_kernel<<<dim3(KBLKS, OUT_F / 256), 256, 0, stream>>>(qw, alpha, beta, wb);

    if (big) {
        x_cvt_kernel<<<(M_ROWS * IN_F) / (256 * 8), 256, 0, stream>>>(x, x16);
        bcq_gemm_kernel<1><<<dim3(M_ROWS / 128, OUT_F / 128), 256, 0, stream>>>(
            nullptr, (const __bf16*)x16, (const __bf16*)wb, y);
    } else {
        bcq_gemm_kernel<0><<<dim3(M_ROWS / 128, OUT_F / 128), 256, 0, stream>>>(
            x, nullptr, (const __bf16*)wb, y);
    }
}
